// CoupFourGAT_84653805404336
// MI455X (gfx1250) — matrix operand order repaired and checked by reading, not compiled, not run on hardware
//
#include <hip/hip_runtime.h>

#define kB 128
#define kN 247
#define kL 12
#define kH 4
#define kXLEN 2964   /* kN*kL */

typedef __attribute__((ext_vector_type(16))) __bf16 v16bf;
typedef __attribute__((ext_vector_type(8)))  float  v8f;

union ABReg { v16bf v; unsigned int u[8]; };

__device__ __forceinline__ unsigned short f2bf(float f) {
  unsigned int u = __float_as_uint(f);
  unsigned int r = (u + 0x7FFFu + ((u >> 16) & 1u)) >> 16;
  return (unsigned short)r;
}

// ---------------------------------------------------------------------------
// Stage 0: additive attention-bias matrix (padded to 256x256):
//   0       where edge present
//  -1e9     where adj == 0          (matches reference mask value)
//  -3e38    on padded rows/cols     (exp -> 0 exactly)
// ---------------------------------------------------------------------------
__global__ __launch_bounds__(256) void bias_kernel(const int* __restrict__ adj,
                                                   float* __restrict__ biasM) {
  int idx = blockIdx.x * blockDim.x + threadIdx.x;
  if (idx >= 256 * 256) return;
  int cg = idx & 255;
  int rg = idx >> 8;
  float v;
  if (rg < kN && cg < kN)
    v = (adj[rg * kN + cg] != 0) ? 0.f : -1.0e9f;
  else
    v = -3.0e38f;
  biasM[idx] = v;
}

// ---------------------------------------------------------------------------
// Stage 1: direct rDFT of reflect-padded, framed signal.
// Writes spec as [b][k(129)][f(12)] (already "swapaxes + [:12]").
// ---------------------------------------------------------------------------
__global__ __launch_bounds__(256) void dft_kernel(const float* __restrict__ x,
                                                  float* __restrict__ sr,
                                                  float* __restrict__ si) {
  int idx = blockIdx.x * blockDim.x + threadIdx.x;
  if (idx >= kB * 129 * 12) return;
  int f = idx % 12;
  int rem = idx / 12;
  int k = rem % 129;
  int b = rem / 129;

  float ang = 6.28318530717958647692f * (float)k * (1.0f / 256.0f);
  float cw = cosf(ang), sw = sinf(ang);
  float c = 1.f, s = 0.f;          // e^{-i*2pi*k*n/256}
  float cr = 0.f, ci = 0.f;
  const float* xb = x + (size_t)b * kXLEN;
  int p0 = f * 246 - 128;          // frame start in unpadded coords
  #pragma unroll 4
  for (int nn = 0; nn < 256; ++nn) {
    int src = p0 + nn;
    if (src < 0) src = -src;
    if (src >= kXLEN) src = 2 * kXLEN - 2 - src;
    float vx = xb[src];
    cr += vx * c;
    ci += vx * s;
    float cn = c * cw + s * sw;
    s = s * cw - c * sw;
    c = cn;
  }
  sr[(b * 129 + k) * 12 + f] = cr;
  si[(b * 129 + k) * 12 + f] = ci;
}

// ---------------------------------------------------------------------------
// Stage 2: reshape (B,129,12)->(B,12,129) [flat reinterpret], @ map_w.T + b,
// then flat-store so the (B,12,247)->(B,247,12) reinterpret is free.
// ---------------------------------------------------------------------------
__global__ __launch_bounds__(256) void map_kernel(const float* __restrict__ sr,
                                                  const float* __restrict__ si,
                                                  const float* __restrict__ map_w,
                                                  const float* __restrict__ map_b,
                                                  float* __restrict__ xrm,
                                                  float* __restrict__ xim) {
  int idx = blockIdx.x * blockDim.x + threadIdx.x;
  if (idx >= kB * 12 * kN) return;
  int nn = idx % kN;
  int rem = idx / kN;
  int i = rem % 12;
  int b = rem / 12;

  float ar = 0.f, ai = 0.f;
  const float* srb = sr + (size_t)b * 129 * 12;
  const float* sib = si + (size_t)b * 129 * 12;
  for (int j = 0; j < 129; ++j) {
    int flat = i * 129 + j;       // index into (129,12) flat
    int r = flat / 12;
    int cc = flat - r * 12;
    float w = map_w[nn * 129 + j];
    ar += srb[r * 12 + cc] * w;
    ai += sib[r * 12 + cc] * w;
  }
  int q = i * kN + nn;            // flat position in (12,247) == (247,12) flat
  xrm[(size_t)b * kXLEN + q] = ar + map_b[nn];
  xim[(size_t)b * kXLEN + q] = ai + map_b[nn];
}

// ---------------------------------------------------------------------------
// Stage 3: 2->2 ch 3x3 conv over (N,L) with the reference's reinterprets,
// leaky-relu + residual; outputs xr2/xi2 in (B,N,L) flat layout.
// ---------------------------------------------------------------------------
__global__ __launch_bounds__(256) void conv_kernel(const float* __restrict__ xrm,
                                                   const float* __restrict__ xim,
                                                   const float* __restrict__ conv_w,
                                                   const float* __restrict__ conv_b,
                                                   float* __restrict__ xr2,
                                                   float* __restrict__ xi2) {
  int idx = blockIdx.x * blockDim.x + threadIdx.x;
  if (idx >= kB * 2 * kXLEN) return;
  int q = idx % (2 * kXLEN);
  int b = idx / (2 * kXLEN);
  int co = q / kXLEN;
  int rq = q - co * kXLEN;
  int n = rq / kL;
  int l = rq - n * kL;

  const float* base[2] = {xrm + (size_t)b * kXLEN, xim + (size_t)b * kXLEN};
  float sum = conv_b[co];
  #pragma unroll
  for (int ci = 0; ci < 2; ++ci) {
    #pragma unroll
    for (int dy = -1; dy <= 1; ++dy) {
      #pragma unroll
      for (int dx = -1; dx <= 1; ++dx) {
        int nn = n + dy, ll = l + dx;
        if (nn < 0 || nn >= kN || ll < 0 || ll >= kL) continue;
        int p = ci * kXLEN + nn * kL + ll;   // flat into (2,N,L) == (N,L,2) reinterpret
        int nv = p / 24;
        int r2 = p - nv * 24;
        int lv = r2 >> 1;
        int tv = r2 & 1;
        float val = base[tv][nv * kL + lv];
        sum += conv_w[((co * 2 + ci) * 3 + (dy + 1)) * 3 + (dx + 1)] * val;
      }
    }
  }
  float res = (sum >= 0.f) ? sum : 0.01f * sum;  // leaky
  // conv-out flat q reinterpreted as (N,L,2), + vec residual at same spot
  int n3 = q / 24;
  int r3 = q - n3 * 24;
  int l3 = r3 >> 1;
  int t3 = r3 & 1;
  float vres = base[t3][n3 * kL + l3];
  float outv = res + vres;
  if (t3 == 0) xr2[(size_t)b * kXLEN + n3 * kL + l3] = outv;
  else         xi2[(size_t)b * kXLEN + n3 * kL + l3] = outv;
}

// ---------------------------------------------------------------------------
// Stage 4: Q/K/V projections -> bf16 WMMA-friendly layouts.
// Qb : [hb][n(256)][32]    A-operand rows, K-dim zero padded 12->32
// Ktb: [hb][o(32)][n(256)] B-operand rows (K^T, lane==K row)
// Vb : [hb][n(256)][16]    B-operand rows for P@V; col 12 = 1.0 so the WMMA
//                          accumulates the softmax denominator for free.
// ---------------------------------------------------------------------------
__global__ __launch_bounds__(256) void qkv_kernel(const float* __restrict__ tin,
                                                  const float* __restrict__ Wq,
                                                  const float* __restrict__ Wk,
                                                  const float* __restrict__ Wv,
                                                  unsigned short* __restrict__ Qb,
                                                  unsigned short* __restrict__ Ktb,
                                                  unsigned short* __restrict__ Vb) {
  int idx = blockIdx.x * blockDim.x + threadIdx.x;
  if (idx >= kH * kB * 256) return;
  int n = idx & 255;
  int hb = idx >> 8;               // h*kB + b
  int h = hb / kB;
  int b = hb - h * kB;

  float q[13], k[12], v[13];
  #pragma unroll
  for (int o = 0; o < 12; ++o) { q[o] = 0.f; k[o] = 0.f; v[o] = 0.f; }
  q[12] = 0.f;
  v[12] = (n < kN) ? 1.0f : 0.f;   // ones column -> row-sum accumulator
  if (n < kN) {
    const float* t = tin + ((size_t)b * kN + n) * kL;
    #pragma unroll
    for (int l = 0; l < 12; ++l) {
      float tv = t[l];
      #pragma unroll
      for (int o = 0; o < 12; ++o) {
        q[o] += tv * Wq[(h * 12 + l) * 12 + o];
        k[o] += tv * Wk[(h * 12 + l) * 12 + o];
        v[o] += tv * Wv[(h * 12 + l) * 12 + o];
      }
    }
  }
  // Q row (32 bf16 = 16 dwords)
  unsigned int* qr = (unsigned int*)(Qb + (((size_t)hb * 256) + n) * 32);
  #pragma unroll
  for (int dw = 0; dw < 16; ++dw) {
    int e0 = 2 * dw, e1 = 2 * dw + 1;
    unsigned int lo = (e0 < 12 && n < kN) ? f2bf(q[e0]) : 0u;
    unsigned int hi = (e1 < 12 && n < kN) ? f2bf(q[e1]) : 0u;
    qr[dw] = lo | (hi << 16);
  }
  // K^T scatter: Kt[o][n]
  unsigned short* kt = Ktb + (size_t)hb * 32 * 256;
  #pragma unroll
  for (int o = 0; o < 32; ++o)
    kt[o * 256 + n] = (o < 12 && n < kN) ? f2bf(k[o]) : (unsigned short)0;
  // V row (16 bf16 = 8 dwords), col 12 = 1.0
  unsigned int* vr = (unsigned int*)(Vb + (((size_t)hb * 256) + n) * 16);
  #pragma unroll
  for (int dw = 0; dw < 8; ++dw) {
    int e0 = 2 * dw, e1 = 2 * dw + 1;
    unsigned int lo = (e0 <= 12 && n < kN) ? f2bf(v[e0]) : 0u;
    unsigned int hi = (e1 <= 12 && n < kN) ? f2bf(v[e1]) : 0u;
    vr[dw] = lo | (hi << 16);
  }
}

// ---------------------------------------------------------------------------
// Stage 5: masked flash-attention + LayerNorm, fully WMMA.
// One block per (h,b); 8 waves x 2 row-tiles of 16 rows.
// Branchless masking via additive bias; softmax denominator rides in
// accumulator column 12 (ones-column of V); butterfly reductions are
// level-ordered so the 8 ds_bpermutes per level are independent.
// ---------------------------------------------------------------------------
__global__ __launch_bounds__(256) void attn_kernel(const unsigned short* __restrict__ Qb,
                                                   const unsigned short* __restrict__ Ktb,
                                                   const unsigned short* __restrict__ Vb,
                                                   const float* __restrict__ biasM,
                                                   const float* __restrict__ ln_g,
                                                   const float* __restrict__ ln_b,
                                                   float* __restrict__ aout) {
  __shared__ unsigned short Kt[32][256];   // 16 KB: B-operand rows for scores
  __shared__ unsigned short Vt[256][16];   //  8 KB: B-operand rows for P@V
  __shared__ unsigned short Pb[8][16][32]; //  8 KB: per-wave P staging (A layout src)

  const int hb = blockIdx.x;               // h*kB + b
  const int tid = threadIdx.x;
  const int wave = tid >> 5;
  const int lane = tid & 31;
  const int half = lane >> 4;
  const int col = lane & 15;

  {  // cooperative staging of K^T and V for this (h,b)
    const unsigned int* ksrc = (const unsigned int*)(Ktb + (size_t)hb * 32 * 256);
    unsigned int* kdst = (unsigned int*)(&Kt[0][0]);
    for (int i = tid; i < 32 * 256 / 2; i += 256) kdst[i] = ksrc[i];
    const unsigned int* vsrc = (const unsigned int*)(Vb + (size_t)hb * 256 * 16);
    unsigned int* vdst = (unsigned int*)(&Vt[0][0]);
    for (int i = tid; i < 256 * 16 / 2; i += 256) vdst[i] = vsrc[i];
  }
  // zero the K=16..31 half of this wave's P staging buffer (once)
  for (int i = lane; i < 16 * 8; i += 32) {
    int r = i >> 3, c = i & 7;
    ((unsigned int*)(&Pb[wave][r][16]))[c] = 0u;
  }
  __syncthreads();

  const float lgc = (col < 12) ? ln_g[col] : 0.f;
  const float lbc = (col < 12) ? ln_b[col] : 0.f;
  const float scale = 0.28867513459481287f;  // 1/sqrt(12)

  for (int it = 0; it < 2; ++it) {
    const int i0 = (wave * 2 + it) * 16;
    // A operand: Q rows i0..i0+15, per-lane gather into ISA 16-bit A(16x32) layout
    ABReg aq;
    {
      const unsigned int* qrow =
          (const unsigned int*)(Qb + (((size_t)hb * 256) + (i0 + col)) * 32);
      #pragma unroll
      for (int v = 0; v < 8; ++v) {
        int dw = v + (v >= 4 ? 4 : 0) + 4 * half;
        aq.u[v] = qrow[dw];
      }
    }

    v8f acc = {};
    float m[8];
    #pragma unroll
    for (int r = 0; r < 8; ++r) m[r] = -3.0e38f;

    // bias row base for this tile's rows (rows i0 + 8*half + r)
    const float* bbase = biasM + ((size_t)(i0 + 8 * half) << 8) + col;

    for (int j = 0; j < 16; ++j) {
      const int j0 = j * 16;
      // issue the 8 independent bias loads early (unconditional, L2-hot)
      float bias[8];
      #pragma unroll
      for (int r = 0; r < 8; ++r) bias[r] = bbase[(r << 8) + j0];

      // B operand: K^T column j0+col, K rows gathered in table order
      ABReg bk;
      {
        #pragma unroll
        for (int v = 0; v < 8; ++v) {
          int k0 = 2 * v + (v >= 4 ? 8 : 0) + 8 * half;
          bk.u[v] = (unsigned int)Kt[k0][j0 + col] | ((unsigned int)Kt[k0 + 1][j0 + col] << 16);
        }
      }
      v8f Sm = {};
      Sm = __builtin_amdgcn_wmma_f32_16x16x32_bf16(false, aq.v, false, bk.v,
                                                   (short)0, Sm, false, false);
      // branchless scale + mask
      float t[8];
      #pragma unroll
      for (int r = 0; r < 8; ++r) {
        float sv = fmaf(Sm[r], scale, bias[r]);
        Sm[r] = sv;
        t[r] = sv;
      }
      // level-ordered row-max butterfly (8 independent shuffles per level)
      #pragma unroll
      for (int d = 1; d < 16; d <<= 1) {
        #pragma unroll
        for (int r = 0; r < 8; ++r)
          t[r] = fmaxf(t[r], __shfl_xor(t[r], d, 16));
      }
      // online rescale + exp (denominator lives in acc column 12)
      v8f P;
      #pragma unroll
      for (int r = 0; r < 8; ++r) {
        float mn = fmaxf(m[r], t[r]);
        float fs = __expf(m[r] - mn);
        acc[r] *= fs;
        m[r] = mn;
        P[r] = __expf(Sm[r] - mn);
      }
      // restage P through LDS: C/D layout -> A(16x32) layout (cols 16..31 stay 0)
      #pragma unroll
      for (int r = 0; r < 8; ++r)
        Pb[wave][r + 8 * half][col] = f2bf(P[r]);
      asm volatile("s_wait_dscnt 0" ::: "memory");
      ABReg ap;
      {
        const unsigned int* pr = (const unsigned int*)(&Pb[wave][col][0]);
        #pragma unroll
        for (int v = 0; v < 8; ++v) {
          int dw = v + (v >= 4 ? 4 : 0) + 4 * half;
          ap.u[v] = pr[dw];
        }
      }
      // B operand: V column col, rows j0+k in table order (k >= 16 multiply zero P columns)
      ABReg bv;
      {
        #pragma unroll
        for (int v = 0; v < 8; ++v) {
          int k0 = 2 * v + (v >= 4 ? 8 : 0) + 8 * half;
          int vr0 = j0 + k0, vr1 = j0 + k0 + 1;
          if (vr0 > 255) vr0 = 255;
          if (vr1 > 255) vr1 = 255;
          bv.u[v] = (unsigned int)Vt[vr0][col] | ((unsigned int)Vt[vr1][col] << 16);
        }
      }
      acc = __builtin_amdgcn_wmma_f32_16x16x32_bf16(false, ap.v, false, bv.v,
                                                    (short)0, acc, false, false);
    }

    // normalize by the ones-column denominator, LayerNorm over L=12, store
    float xn[8];
    #pragma unroll
    for (int r = 0; r < 8; ++r) {
      float srow = __shfl(acc[r], 12, 16);   // broadcast denominator
      xn[r] = acc[r] / srow;
    }
    float sm[8];
    #pragma unroll
    for (int r = 0; r < 8; ++r) sm[r] = (col < 12) ? xn[r] : 0.f;
    #pragma unroll
    for (int d = 1; d < 16; d <<= 1) {
      #pragma unroll
      for (int r = 0; r < 8; ++r) sm[r] += __shfl_xor(sm[r], d, 16);
    }
    float mean[8], dv[8];
    #pragma unroll
    for (int r = 0; r < 8; ++r) {
      mean[r] = sm[r] * (1.f / 12.f);
      float d0 = (col < 12) ? (xn[r] - mean[r]) : 0.f;
      dv[r] = d0 * d0;
    }
    #pragma unroll
    for (int d = 1; d < 16; d <<= 1) {
      #pragma unroll
      for (int r = 0; r < 8; ++r) dv[r] += __shfl_xor(dv[r], d, 16);
    }
    #pragma unroll
    for (int r = 0; r < 8; ++r) {
      float inv = __frsqrt_rn(dv[r] * (1.f / 12.f) + 1e-5f);
      float y = (xn[r] - mean[r]) * inv * lgc + lbc;
      int rg = i0 + r + 8 * half;
      if (rg < kN && col < 12)
        aout[((size_t)hb * kN + rg) * kL + col] = y;
    }
  }
}

// ---------------------------------------------------------------------------
// Stage 6: gated fusion MLP (with the reference's reshape scrambling),
// + head-mean, softshrink; writes (B,N,L).
// ---------------------------------------------------------------------------
__global__ __launch_bounds__(256) void fuse_kernel(const float* __restrict__ attn,
                                                   const float* __restrict__ g1_w,
                                                   const float* __restrict__ g1_b,
                                                   const float* __restrict__ gln_g,
                                                   const float* __restrict__ gln_b,
                                                   const float* __restrict__ g2_w,
                                                   const float* __restrict__ g2_b,
                                                   float* __restrict__ dst) {
  int idx = blockIdx.x * blockDim.x + threadIdx.x;
  if (idx >= kN * kB) return;
  int n = idx / kB;
  int b3 = idx - n * kB;   // i3 axis of nfr

  float avg[12], mx[12];
  #pragma unroll
  for (int l2 = 0; l2 < 12; ++l2) {
    float a = 0.f, mv = -3.0e38f;
    #pragma unroll
    for (int i2 = 0; i2 < 4; ++i2) {
      int fidx = i2 * 1536 + b3 * 12 + l2;   // flat in (B,H,L) block of nf
      int bs = fidx / 48;
      int rm = fidx - bs * 48;
      int hs = rm / 12;
      int ls = rm - hs * 12;
      float val = attn[(((size_t)hs * kB + bs) * kN + n) * kL + ls];
      a += val;
      mv = fmaxf(mv, val);
    }
    avg[l2] = a * 0.25f;
    mx[l2] = mv;
  }
  // z = [avg, mx] @ g1_w.T + g1_b
  float z[12];
  #pragma unroll
  for (int o = 0; o < 12; ++o) {
    float acc = g1_b[o];
    #pragma unroll
    for (int j = 0; j < 12; ++j) {
      acc += avg[j] * g1_w[o * 24 + j];
      acc += mx[j] * g1_w[o * 24 + 12 + j];
    }
    z[o] = acc;
  }
  // LayerNorm
  float mean = 0.f;
  #pragma unroll
  for (int o = 0; o < 12; ++o) mean += z[o];
  mean *= (1.f / 12.f);
  float var = 0.f;
  #pragma unroll
  for (int o = 0; o < 12; ++o) { float d = z[o] - mean; var += d * d; }
  var *= (1.f / 12.f);
  float inv = __frsqrt_rn(var + 1e-5f);
  #pragma unroll
  for (int o = 0; o < 12; ++o) {
    float zn = (z[o] - mean) * inv * gln_g[o] + gln_b[o];
    z[o] = 0.5f * zn * (1.f + erff(zn * 0.70710678118654752f));  // exact gelu
  }
  // sigmoid gate, combine, + mean over real heads, softshrink
  #pragma unroll
  for (int o = 0; o < 12; ++o) {
    float acc = g2_b[o];
    #pragma unroll
    for (int j = 0; j < 12; ++j) acc += z[j] * g2_w[o * 12 + j];
    float zz = 1.f / (1.f + __expf(-acc));
    float fused = zz * avg[o] + (1.f - zz) * mx[o];
    float mean2 = 0.f;
    #pragma unroll
    for (int h = 0; h < kH; ++h)
      mean2 += attn[(((size_t)h * kB + b3) * kN + n) * kL + o];
    float v = fused + mean2 * 0.25f;
    v = (v > 0.01f) ? (v - 0.01f) : ((v < -0.01f) ? (v + 0.01f) : 0.f);  // softshrink
    dst[((size_t)b3 * kN + n) * kL + o] = v;
  }
}

// ---------------------------------------------------------------------------
// Stage 7: combine xr/xi, LN(ln2) + residual, encode, leaky, decode.
// ---------------------------------------------------------------------------
__global__ __launch_bounds__(256) void final_kernel(const float* __restrict__ ar,
                                                    const float* __restrict__ ai,
                                                    const float* __restrict__ x,
                                                    const float* __restrict__ prc,
                                                    const float* __restrict__ enc_w,
                                                    const float* __restrict__ enc_b,
                                                    const float* __restrict__ ln2_g,
                                                    const float* __restrict__ ln2_b,
                                                    const float* __restrict__ dec_w,
                                                    const float* __restrict__ dec_b,
                                                    float* __restrict__ out) {
  int idx = blockIdx.x * blockDim.x + threadIdx.x;
  if (idx >= kB * kN) return;
  size_t base = (size_t)idx * kL;
  float e00 = enc_w[0], e01 = enc_w[1], eb = enc_b[0];

  float g[12];
  #pragma unroll
  for (int l = 0; l < 12; ++l) g[l] = ar[base + l] * e00 + ai[base + l] * e01 + eb;
  // LN over L with ln2
  float mean = 0.f;
  #pragma unroll
  for (int l = 0; l < 12; ++l) mean += g[l];
  mean *= (1.f / 12.f);
  float var = 0.f;
  #pragma unroll
  for (int l = 0; l < 12; ++l) { float d = g[l] - mean; var += d * d; }
  var *= (1.f / 12.f);
  float inv = __frsqrt_rn(var + 1e-5f);
  float a[12];
  #pragma unroll
  for (int l = 0; l < 12; ++l) {
    float f = (g[l] - mean) * inv * ln2_g[l] + ln2_b[l] + x[base + l];
    float e = f * e00 + prc[base + l] * e01 + eb;
    a[l] = (e >= 0.f) ? e : 0.01f * e;  // leaky
  }
  #pragma unroll
  for (int l = 0; l < 12; ++l) {
    float acc = dec_b[l] + x[base + l];
    #pragma unroll
    for (int j = 0; j < 12; ++j) acc += a[j] * dec_w[l * 12 + j];
    out[base + l] = acc;
  }
}

// ---------------------------------------------------------------------------
extern "C" void kernel_launch(void* const* d_in, const int* in_sizes, int n_in,
                              void* d_out, int out_size, void* d_ws, size_t ws_size,
                              hipStream_t stream) {
  (void)in_sizes; (void)n_in; (void)out_size; (void)ws_size;
  const float* x     = (const float*)d_in[0];
  const float* prc   = (const float*)d_in[1];
  const int*   adj   = (const int*)d_in[2];
  const float* Wq    = (const float*)d_in[3];
  const float* Wk    = (const float*)d_in[4];
  const float* Wv    = (const float*)d_in[5];
  const float* ln_g  = (const float*)d_in[6];
  const float* ln_b  = (const float*)d_in[7];
  const float* ln2_g = (const float*)d_in[8];
  const float* ln2_b = (const float*)d_in[9];
  const float* enc_w = (const float*)d_in[10];
  const float* enc_b = (const float*)d_in[11];
  const float* dec_w = (const float*)d_in[12];
  const float* dec_b = (const float*)d_in[13];
  const float* map_w = (const float*)d_in[14];
  const float* map_b = (const float*)d_in[15];
  const float* conv_w = (const float*)d_in[16];
  const float* conv_b = (const float*)d_in[17];
  const float* g1_w  = (const float*)d_in[18];
  const float* g1_b  = (const float*)d_in[19];
  const float* gln_g = (const float*)d_in[20];
  const float* gln_b = (const float*)d_in[21];
  const float* g2_w  = (const float*)d_in[22];
  const float* g2_b  = (const float*)d_in[23];

  char* ws = (char*)d_ws;
  size_t off = 0;
  auto alloc = [&](size_t bytes) -> char* {
    char* p = ws + off;
    off = (off + bytes + 255) & ~(size_t)255;
    return p;
  };
  size_t szSpec = (size_t)kB * 129 * 12 * sizeof(float);
  size_t szBNL  = (size_t)kB * kXLEN * sizeof(float);
  float* biasM = (float*)alloc((size_t)256 * 256 * sizeof(float));
  float* specR = (float*)alloc(szSpec);
  float* specI = (float*)alloc(szSpec);
  float* xrm   = (float*)alloc(szBNL);
  float* xim   = (float*)alloc(szBNL);
  float* xr2   = (float*)alloc(szBNL);
  float* xi2   = (float*)alloc(szBNL);
  unsigned short* Qb  = (unsigned short*)alloc((size_t)kH * kB * 256 * 32 * 2);
  unsigned short* Ktb = (unsigned short*)alloc((size_t)kH * kB * 32 * 256 * 2);
  unsigned short* Vb  = (unsigned short*)alloc((size_t)kH * kB * 256 * 16 * 2);
  float* attn  = (float*)alloc((size_t)kH * kB * kN * kL * sizeof(float));
  float* arbuf = (float*)alloc(szBNL);
  float* aibuf = (float*)alloc(szBNL);

  bias_kernel<<<(256 * 256 + 255) / 256, 256, 0, stream>>>(adj, biasM);
  dft_kernel<<<(kB * 129 * 12 + 255) / 256, 256, 0, stream>>>(x, specR, specI);
  map_kernel<<<(kB * 12 * kN + 255) / 256, 256, 0, stream>>>(specR, specI, map_w,
                                                             map_b, xrm, xim);
  conv_kernel<<<(kB * 2 * kXLEN + 255) / 256, 256, 0, stream>>>(xrm, xim, conv_w,
                                                                conv_b, xr2, xi2);
  for (int t = 0; t < 2; ++t) {
    const float* tin = t ? xi2 : xr2;
    qkv_kernel<<<(kH * kB * 256 + 255) / 256, 256, 0, stream>>>(tin, Wq, Wk, Wv,
                                                                Qb, Ktb, Vb);
    attn_kernel<<<kH * kB, 256, 0, stream>>>(Qb, Ktb, Vb, biasM, ln_g, ln_b, attn);
    fuse_kernel<<<(kN * kB + 255) / 256, 256, 0, stream>>>(
        attn, g1_w, g1_b, gln_g, gln_b, g2_w, g2_b, t ? aibuf : arbuf);
  }
  final_kernel<<<(kB * kN + 255) / 256, 256, 0, stream>>>(
      arbuf, aibuf, x, prc, enc_w, enc_b, ln2_g, ln2_b, dec_w, dec_b,
      (float*)d_out);
}
